// GroupQueryAttention_58128087384352
// MI455X (gfx1250) — matrix operand order repaired and checked by reading, not compiled, not run on hardware
//
#include <hip/hip_runtime.h>

// ---------------------------------------------------------------------------
// GQA forward for MI455X (gfx1250, wave32).
//  - All matmuls via V_WMMA_F32_16X16X32_BF16 (bf16 in, f32 accumulate)
//  - Projection GEMMs: LDS block-tiled (128x128x64), double buffered,
//    tiles moved by the Tensor Data Mover (tensor_load_to_lds + TENSORcnt)
//  - Attention: flash-style online softmax, one wave per 16-query tile,
//    P re-layout through LDS with explicit s_wait_dscnt
// ---------------------------------------------------------------------------

typedef __attribute__((ext_vector_type(16))) __bf16 v16bf;
typedef __attribute__((ext_vector_type(8)))  __bf16 v8bf;
typedef __attribute__((ext_vector_type(8)))  float  v8f;
typedef __attribute__((ext_vector_type(4)))  unsigned int v4u;
typedef __attribute__((ext_vector_type(8)))  int v8i;
typedef __attribute__((ext_vector_type(4)))  int v4i;

#define B_     2
#define S_     2048
#define HID_   4096
#define H_     32
#define G_     8
#define D_     128
#define NREP_  4
#define BS_    (B_ * S_)
#define QT_    (S_ / 16)
#define SCALE_ 0.08838834764831845f  // 1/sqrt(128)

// GEMM block tiling
#define BM_ 128
#define BN_ 128
#define BK_ 64

#if defined(__AMDGCN__) && __has_builtin(__builtin_amdgcn_tensor_load_to_lds) && \
    __has_builtin(__builtin_amdgcn_s_wait_tensorcnt)
#define HAVE_TDM 1
#else
#define HAVE_TDM 0
#endif

// fp32 -> bf16 round-to-nearest-even
__device__ __forceinline__ __bf16 f2bf(float f) {
  unsigned u = __builtin_bit_cast(unsigned, f);
  unsigned short h = (unsigned short)((u + 0x7FFFu + ((u >> 16) & 1u)) >> 16);
  return __builtin_bit_cast(__bf16, h);
}

// A-fragment (16-bit 16x32): 16 contiguous K values, elements 0..15 = p[0..15]
__device__ __forceinline__ v16bf frag_ab(const __bf16* p) {
  v8bf lo = *(const v8bf*)p;
  v8bf hi = *(const v8bf*)(p + 8);
  v16bf r;
#pragma unroll
  for (int e = 0; e < 8; ++e) { r[e] = lo[e]; r[e + 8] = hi[e]; }
  return r;
}

// B-fragment (16-bit 32x16): 16 contiguous K values
__device__ __forceinline__ v16bf frag_b16(const __bf16* p) {
  v8bf lo = *(const v8bf*)p;
  v8bf hi = *(const v8bf*)(p + 8);
  v16bf r;
#pragma unroll
  for (int e = 0; e < 8; ++e) { r[e] = lo[e]; r[e + 8] = hi[e]; }
  return r;
}

__device__ __forceinline__ v8f wmma_bf16(v16bf a, v16bf b, v8f c) {
  return __builtin_amdgcn_wmma_f32_16x16x32_bf16(false, a, false, b,
                                                 (short)0, c, false, false);
}

#if HAVE_TDM
// Issue one 2D TDM tile load: rows x BK_ bf16 tile, row-major packed into LDS.
// D# built per CDNA5 ISA ch.8: group0 = {count=1, lds_addr, global_addr, type=2},
// group1 = {data_size=2B, tensor dims, tile dims, dim0 stride}.
__device__ __forceinline__ void tdm_load_2d(unsigned lds_off, const __bf16* gptr,
                                            unsigned tensor_d0, unsigned tensor_d1,
                                            unsigned tile_d0, unsigned tile_d1,
                                            unsigned long long stride0) {
  unsigned long long ga = (unsigned long long)(uintptr_t)gptr;
  unsigned long long g0q0 = 1ull | ((unsigned long long)lds_off << 32);
  unsigned long long g0q1 = (ga & 0x01FFFFFFFFFFFFFFull) | (2ull << 62);
  v4u g0 = {(unsigned)g0q0, (unsigned)(g0q0 >> 32),
            (unsigned)g0q1, (unsigned)(g0q1 >> 32)};
  // group1 bitfields (bit offsets within 256-bit group):
  //  [17:16]=data_size(1->2B)  [79:48]=tensor_dim0  [111:80]=tensor_dim1
  //  [127:112]=tile_dim0  [143:128]=tile_dim1  [207:160]=tensor_dim0_stride
  unsigned long long q0 = (1ull << 16) |
                          ((unsigned long long)(tensor_d0 & 0xFFFFu) << 48);
  unsigned long long q1 = ((unsigned long long)(tensor_d0 >> 16) & 0xFFFFull) |
                          ((unsigned long long)tensor_d1 << 16) |
                          ((unsigned long long)(tile_d0 & 0xFFFFu) << 48);
  unsigned long long q2 = (unsigned long long)(tile_d1 & 0xFFFFu) |
                          ((stride0 & 0xFFFFFFFFull) << 32);
  unsigned long long q3 = (stride0 >> 32) & 0xFFFFull;
  v8i g1 = {(int)q0, (int)(q0 >> 32), (int)q1, (int)(q1 >> 32),
            (int)q2, (int)(q2 >> 32), (int)q3, (int)(q3 >> 32)};
  v4i z4 = {0, 0, 0, 0};
#if __clang_major__ >= 23
  v8i z8 = {0, 0, 0, 0, 0, 0, 0, 0};
  __builtin_amdgcn_tensor_load_to_lds(g0, g1, z4, z4, z8, 0);
#else
  __builtin_amdgcn_tensor_load_to_lds(g0, g1, z4, z4, 0);
#endif
}
#endif  // HAVE_TDM

// ---------------------------------------------------------------------------
__global__ void cvt_f32_bf16(const float* __restrict__ src,
                             __bf16* __restrict__ dst, int n) {
  int i = blockIdx.x * blockDim.x + threadIdx.x;
  if (i < n) dst[i] = f2bf(src[i]);
}

// src[K][N] fp32 -> dst[N][K] bf16 (so WMMA B-fragments are contiguous)
__global__ void transpose_cvt(const float* __restrict__ src,
                              __bf16* __restrict__ dst, int K, int N) {
  int i = blockIdx.x * blockDim.x + threadIdx.x;
  if (i < K * N) {
    int k = i / N, n = i - k * N;
    dst[(size_t)n * K + k] = f2bf(src[i]);
  }
}

// ---------------------------------------------------------------------------
// C[M][N] f32 = A[M][K] bf16 x BT[N][K] bf16.  Block: 256 thr (8 waves),
// 128x128 tile, K stepped by 64, double-buffered LDS fed by the TDM.
// Wave (wr,wc) in 4x2 grid computes a 32x64 sub-tile (2x4 WMMA accumulators).
__global__ __launch_bounds__(256) void gemm_tiled(
    const __bf16* __restrict__ A, const __bf16* __restrict__ BT,
    float* __restrict__ C, int M, int N, int K) {
  extern __shared__ __bf16 smem[];
  __bf16* sA = smem;                      // [2][BM_*BK_]
  __bf16* sB = smem + 2 * BM_ * BK_;      // [2][BN_*BK_]

  int nbn = N / BN_;
  int bm = blockIdx.x / nbn, bn = blockIdx.x - bm * nbn;
  int lane = threadIdx.x & 31, wslot = threadIdx.x >> 5;
  int l16 = lane & 15, halo = lane >> 4;
  int wr = wslot >> 1, wc = wslot & 1;
  int nk = K / BK_;

  const __bf16* Abase = A + (size_t)bm * BM_ * K;
  const __bf16* Bbase = BT + (size_t)bn * BN_ * K;

  v8f acc[2][4];
  v8f z = {};
#pragma unroll
  for (int mt = 0; mt < 2; ++mt)
#pragma unroll
    for (int nt = 0; nt < 4; ++nt) acc[mt][nt] = z;

  auto stage_load = [&](int kt, int buf) {
#if HAVE_TDM
    if (threadIdx.x < 32) {  // wave 0 drives the DMA (TDM ignores EXEC)
      tdm_load_2d((unsigned)(uintptr_t)(sA + buf * (BM_ * BK_)),
                  Abase + (size_t)kt * BK_, (unsigned)K, 1u << 20,
                  BK_, BM_, (unsigned long long)K);
      tdm_load_2d((unsigned)(uintptr_t)(sB + buf * (BN_ * BK_)),
                  Bbase + (size_t)kt * BK_, (unsigned)K, 1u << 20,
                  BK_, BN_, (unsigned long long)K);
    }
#else
    // cooperative fallback: 16B chunks, all 256 threads
    {
      __bf16* dA = sA + buf * (BM_ * BK_);
      for (int idx = threadIdx.x; idx < BM_ * (BK_ / 8); idx += 256) {
        int r = idx >> 3, c = idx & 7;
        *(v8bf*)(dA + r * BK_ + c * 8) =
            *(const v8bf*)(Abase + (size_t)r * K + (size_t)kt * BK_ + c * 8);
      }
      __bf16* dB = sB + buf * (BN_ * BK_);
      for (int idx = threadIdx.x; idx < BN_ * (BK_ / 8); idx += 256) {
        int r = idx >> 3, c = idx & 7;
        *(v8bf*)(dB + r * BK_ + c * 8) =
            *(const v8bf*)(Bbase + (size_t)r * K + (size_t)kt * BK_ + c * 8);
      }
    }
#endif
  };

  stage_load(0, 0);
  for (int kt = 0; kt < nk; ++kt) {
    int buf = kt & 1;
    bool more = (kt + 1 < nk);
    if (more) stage_load(kt + 1, buf ^ 1);
#if HAVE_TDM
    if (threadIdx.x < 32) {
      if (more) __builtin_amdgcn_s_wait_tensorcnt(2);  // current stage done
      else      __builtin_amdgcn_s_wait_tensorcnt(0);
    }
#endif
    __syncthreads();

    const __bf16* As = sA + buf * (BM_ * BK_);
    const __bf16* Bs = sB + buf * (BN_ * BK_);
#pragma unroll
    for (int kk = 0; kk < BK_; kk += 32) {
      v16bf am0 = frag_ab(As + (wr * 32 + l16) * BK_ + kk + halo * 16);
      v16bf am1 = frag_ab(As + (wr * 32 + 16 + l16) * BK_ + kk + halo * 16);
#pragma unroll
      for (int nt = 0; nt < 4; ++nt) {
        v16bf bf = frag_b16(Bs + (wc * 64 + nt * 16 + l16) * BK_ + kk + halo * 16);
        acc[0][nt] = wmma_bf16(am0, bf, acc[0][nt]);
        acc[1][nt] = wmma_bf16(am1, bf, acc[1][nt]);
      }
    }
    __syncthreads();
  }

#pragma unroll
  for (int mt = 0; mt < 2; ++mt)
#pragma unroll
    for (int nt = 0; nt < 4; ++nt)
#pragma unroll
      for (int i = 0; i < 8; ++i) {
        int m = bm * BM_ + wr * 32 + mt * 16 + halo * 8 + i;
        int n = bn * BN_ + wc * 64 + nt * 16 + l16;
        C[(size_t)m * N + n] = acc[mt][nt][i];
      }
}

// ---------------------------------------------------------------------------
// RoPE + repack: raw[b*S+s][h*D+d] f32 -> dst[b][h][s][d] bf16
__global__ void rope_pack(const float* __restrict__ raw,
                          const float* __restrict__ cosT,
                          const float* __restrict__ sinT,
                          __bf16* __restrict__ dst, int NH) {
  int i = blockIdx.x * blockDim.x + threadIdx.x;
  int total = B_ * S_ * NH * (D_ / 2);
  if (i >= total) return;
  int p = i % (D_ / 2);
  int h = (i / (D_ / 2)) % NH;
  int s = (i / ((D_ / 2) * NH)) % S_;
  int b = i / ((D_ / 2) * NH * S_);
  size_t si = ((size_t)(b * S_ + s) * NH + h) * D_ + 2 * p;
  float xr = raw[si], xi = raw[si + 1];
  float c = cosT[s * (D_ / 2) + p], sn = sinT[s * (D_ / 2) + p];
  size_t di = (((size_t)(b * NH + h) * S_ + s) * D_) + 2 * p;
  dst[di]     = f2bf(xr * c - xi * sn);
  dst[di + 1] = f2bf(xr * sn + xi * c);
}

// V repack transposed: raw[b*S+s][g*D+d] f32 -> dst[b][g][d][s] bf16
__global__ void v_pack(const float* __restrict__ raw, __bf16* __restrict__ dst) {
  int i = blockIdx.x * blockDim.x + threadIdx.x;
  int total = B_ * S_ * G_ * D_;
  if (i >= total) return;
  int d = i % D_;
  int g = (i / D_) % G_;
  int s = (i / (D_ * G_)) % S_;
  int b = i / (D_ * G_ * S_);
  dst[((size_t)(b * G_ + g) * D_ + d) * S_ + s] = f2bf(raw[i]);
}

// ---------------------------------------------------------------------------
// Flash attention: one wave per (b, h, 16-query tile). Causal, start_pos=0.
// qb[b][h][s][D], kb[b][g][s][D], vT[b][g][D][s]; out: attn[b*S+s][h*D+d] bf16
__global__ __launch_bounds__(128) void attn_kernel(
    const __bf16* __restrict__ qb, const __bf16* __restrict__ kb,
    const __bf16* __restrict__ vT, __bf16* __restrict__ attn) {
  __shared__ __align__(32) __bf16 lds_p[4][16 * 32];
  int lane = threadIdx.x & 31;
  int wslot = threadIdx.x >> 5;
  int w = blockIdx.x * 4 + wslot;
  int qt = w % QT_;
  int h = (w / QT_) % H_;
  int b = w / (QT_ * H_);
  int g = h / NREP_;
  int l16 = lane & 15, halo = lane >> 4;

  const __bf16* qrow = qb + (((size_t)(b * H_ + h) * S_ + qt * 16 + l16) * D_);
  v16bf qf[4];
#pragma unroll
  for (int c = 0; c < 4; ++c) qf[c] = frag_ab(qrow + c * 32 + halo * 16);

  float rmax[8], rsum[8];
  v8f o[8];
  v8f z = {};
#pragma unroll
  for (int i = 0; i < 8; ++i) { rmax[i] = -3e38f; rsum[i] = 0.f; }
#pragma unroll
  for (int j = 0; j < 8; ++j) o[j] = z;

  const __bf16* kbase = kb + ((size_t)(b * G_ + g) * S_) * D_;
  const __bf16* vbase = vT + ((size_t)(b * G_ + g) * D_) * S_;
  __bf16* lp = &lds_p[wslot][0];

  for (int k0 = 0; k0 < (qt + 1) * 16; k0 += 32) {
    v8f s0 = z, s1 = z;
#pragma unroll
    for (int c = 0; c < 4; ++c)
      s0 = wmma_bf16(qf[c],
                     frag_b16(kbase + (size_t)(k0 + l16) * D_ + c * 32 + halo * 16),
                     s0);
#pragma unroll
    for (int c = 0; c < 4; ++c)
      s1 = wmma_bf16(qf[c],
                     frag_b16(kbase + (size_t)(k0 + 16 + l16) * D_ + c * 32 + halo * 16),
                     s1);

    float p0[8], p1[8], red[8];
#pragma unroll
    for (int i = 0; i < 8; ++i) {
      int qrowi = qt * 16 + halo * 8 + i;
      float a0 = s0[i] * SCALE_ + ((k0 + l16) <= qrowi ? 0.f : -1e9f);
      float a1 = s1[i] * SCALE_ + ((k0 + 16 + l16) <= qrowi ? 0.f : -1e9f);
      p0[i] = a0; p1[i] = a1;
      red[i] = fmaxf(a0, a1);
    }
#pragma unroll
    for (int i = 0; i < 8; ++i)
#pragma unroll
      for (int off = 1; off < 16; off <<= 1)
        red[i] = fmaxf(red[i], __shfl_xor(red[i], off, 32));

    float alpha[8];
#pragma unroll
    for (int i = 0; i < 8; ++i) {
      float nm = fmaxf(rmax[i], red[i]);
      alpha[i] = __expf(rmax[i] - nm);
      rmax[i] = nm;
      p0[i] = __expf(p0[i] - nm);
      p1[i] = __expf(p1[i] - nm);
      red[i] = p0[i] + p1[i];
    }
#pragma unroll
    for (int i = 0; i < 8; ++i)
#pragma unroll
      for (int off = 1; off < 16; off <<= 1)
        red[i] += __shfl_xor(red[i], off, 32);
#pragma unroll
    for (int i = 0; i < 8; ++i) rsum[i] = rsum[i] * alpha[i] + red[i];
#pragma unroll
    for (int j = 0; j < 8; ++j)
#pragma unroll
      for (int i = 0; i < 8; ++i) o[j][i] *= alpha[i];

#pragma unroll
    for (int i = 0; i < 8; ++i) {
      lp[(halo * 8 + i) * 32 + l16]      = f2bf(p0[i]);
      lp[(halo * 8 + i) * 32 + 16 + l16] = f2bf(p1[i]);
    }
    asm volatile("s_wait_dscnt 0" ::: "memory");
    v16bf pf = frag_ab(lp + l16 * 32 + halo * 16);

#pragma unroll
    for (int j = 0; j < 8; ++j)
      o[j] = wmma_bf16(pf,
                       frag_b16(vbase + (size_t)(j * 16 + l16) * S_ + k0 + halo * 16),
                       o[j]);
  }

#pragma unroll
  for (int j = 0; j < 8; ++j)
#pragma unroll
    for (int i = 0; i < 8; ++i) {
      float val = o[j][i] / rsum[i];
      size_t row = (size_t)b * S_ + qt * 16 + halo * 8 + i;
      attn[row * (H_ * D_) + h * D_ + j * 16 + l16] = f2bf(val);
    }
}

// ---------------------------------------------------------------------------
extern "C" void kernel_launch(void* const* d_in, const int* in_sizes, int n_in,
                              void* d_out, int out_size, void* d_ws,
                              size_t ws_size, hipStream_t stream) {
  (void)in_sizes; (void)n_in; (void)out_size; (void)ws_size;
  const float* x    = (const float*)d_in[0];
  const float* wq   = (const float*)d_in[1];
  const float* wk   = (const float*)d_in[2];
  const float* wv   = (const float*)d_in[3];
  const float* wo   = (const float*)d_in[4];
  const float* cosT = (const float*)d_in[5];
  const float* sinT = (const float*)d_in[6];
  // d_in[7] = mask (applied arithmetically), d_in[8] = start_pos (== 0)

  char* ws = (char*)d_ws;
  size_t off = 0;
  auto take = [&](size_t bytes) -> char* {
    char* p = ws + off;
    off += (bytes + 255) & ~(size_t)255;
    return p;
  };

  __bf16* xb   = (__bf16*)take((size_t)BS_ * HID_ * 2);
  __bf16* wqT  = (__bf16*)take((size_t)HID_ * HID_ * 2);
  __bf16* wkT  = (__bf16*)take((size_t)(G_ * D_) * HID_ * 2);
  __bf16* wvT  = (__bf16*)take((size_t)(G_ * D_) * HID_ * 2);
  __bf16* woT  = (__bf16*)take((size_t)HID_ * HID_ * 2);
  float*  qraw = (float*)take((size_t)BS_ * (H_ * D_) * 4);
  float*  kraw = (float*)take((size_t)BS_ * (G_ * D_) * 4);
  float*  vraw = (float*)take((size_t)BS_ * (G_ * D_) * 4);
  __bf16* qbuf = (__bf16*)take((size_t)B_ * H_ * S_ * D_ * 2);
  __bf16* kbuf = (__bf16*)take((size_t)B_ * G_ * S_ * D_ * 2);
  __bf16* vbuf = (__bf16*)take((size_t)B_ * G_ * D_ * S_ * 2);
  __bf16* attn = (__bf16*)take((size_t)BS_ * (H_ * D_) * 2);

  // 1) fp32 -> bf16 conversions (+ weight transposes)
  {
    int n = BS_ * HID_;
    cvt_f32_bf16<<<(n + 255) / 256, 256, 0, stream>>>(x, xb, n);
  }
  transpose_cvt<<<(HID_ * (H_ * D_) + 255) / 256, 256, 0, stream>>>(wq, wqT, HID_, H_ * D_);
  transpose_cvt<<<(HID_ * (G_ * D_) + 255) / 256, 256, 0, stream>>>(wk, wkT, HID_, G_ * D_);
  transpose_cvt<<<(HID_ * (G_ * D_) + 255) / 256, 256, 0, stream>>>(wv, wvT, HID_, G_ * D_);
  transpose_cvt<<<((H_ * D_) * HID_ + 255) / 256, 256, 0, stream>>>(wo, woT, H_ * D_, HID_);

  // 2) TDM-fed LDS-tiled WMMA GEMMs
  const unsigned shmem = 2u * (BM_ * BK_ + BN_ * BK_) * sizeof(__bf16);  // 64 KB
  auto gemm = [&](const __bf16* A, const __bf16* BT, float* C, int M, int N, int K) {
    int blocks = (M / BM_) * (N / BN_);
    gemm_tiled<<<blocks, 256, shmem, stream>>>(A, BT, C, M, N, K);
  };
  gemm(xb, wqT, qraw, BS_, H_ * D_, HID_);
  gemm(xb, wkT, kraw, BS_, G_ * D_, HID_);
  gemm(xb, wvT, vraw, BS_, G_ * D_, HID_);

  // 3) RoPE + layout packs
  {
    int n = B_ * S_ * H_ * (D_ / 2);
    rope_pack<<<(n + 255) / 256, 256, 0, stream>>>(qraw, cosT, sinT, qbuf, H_);
  }
  {
    int n = B_ * S_ * G_ * (D_ / 2);
    rope_pack<<<(n + 255) / 256, 256, 0, stream>>>(kraw, cosT, sinT, kbuf, G_);
  }
  {
    int n = B_ * S_ * G_ * D_;
    v_pack<<<(n + 255) / 256, 256, 0, stream>>>(vraw, vbuf);
  }

  // 4) flash attention: B*H*(S/16) waves, 4 waves/block
  {
    int waves = B_ * H_ * QT_;
    attn_kernel<<<waves / 4, 128, 0, stream>>>(qbuf, kbuf, vbuf, attn);
  }

  // 5) output projection -> d_out (f32)
  gemm(attn, woT, (float*)d_out, BS_, HID_, H_ * D_);
}
